// VectorQuantizer_4561255268795
// MI455X (gfx1250) — matrix operand order repaired and checked by reading, not compiled, not run on hardware
//
#include <hip/hip_runtime.h>
#include <stdint.h>

// ---------- problem constants ----------
#define NUM_EMB    4096
#define EMBED_DIM  256
#define N_ROWS     65536                 // 16 * 4096 flattened rows
#define NZQ        ((size_t)N_ROWS * EMBED_DIM)   // 16,777,216
#define BETA       0.25f

typedef __attribute__((ext_vector_type(16))) __bf16 v16bf;
typedef __attribute__((ext_vector_type(8)))  float  v8f;

union BFrag {
    v16bf     v;
    uint16_t  h[16];
    uint4     u4[2];
};

__device__ __forceinline__ uint16_t f32_to_bf16_bits(float f) {
    union { float f; uint32_t u; } cv; cv.f = f;
    uint32_t u = cv.u;
    // round-to-nearest-even
    uint32_t r = u + 0x7FFFu + ((u >> 16) & 1u);
    return (uint16_t)(r >> 16);
}

// ---------- kernel 0: zero the loss accumulator ----------
__global__ void vq_init(float* __restrict__ accum) {
    if (blockIdx.x == 0 && threadIdx.x == 0) accum[0] = 0.0f;
}

// ---------- kernel 1: emb -> bf16 copy + exact f32 ||e||^2 ----------
__global__ __launch_bounds__(EMBED_DIM)
void vq_prep_emb(const float* __restrict__ emb,
                 uint16_t* __restrict__ emb_bf,
                 float* __restrict__ enorm) {
    __shared__ float red[EMBED_DIM];
    const int k = blockIdx.x;      // code id
    const int d = threadIdx.x;     // dim
    float v = emb[(size_t)k * EMBED_DIM + d];
    emb_bf[(size_t)k * EMBED_DIM + d] = f32_to_bf16_bits(v);
    red[d] = v * v;
    __syncthreads();
    #pragma unroll
    for (int s = EMBED_DIM / 2; s > 0; s >>= 1) {
        if (d < s) red[d] += red[d + s];
        __syncthreads();
    }
    if (d == 0) enorm[k] = red[0];
}

// ---------- kernel 2: bf16-WMMA distance GEMM + per-row argmin ----------
// One wave = 32 rows of z (two 16-row M-tiles sharing every B fragment)
// vs all 4096 codes. B double-buffered; sched_group_barrier pins the
// [loads(k+1)][wmma(k)] interleave so loads stay one step ahead.
__global__ __launch_bounds__(128)
void vq_argmin(const float* __restrict__ z,
               const uint16_t* __restrict__ emb_bf,
               const float* __restrict__ enorm,
               int* __restrict__ best_idx) {
    const int lane     = threadIdx.x & 31;
    const int wave     = threadIdx.x >> 5;
    const int rowStart = (blockIdx.x * 4 + wave) * 32;
    const int m        = lane & 15;   // A-matrix row within tile / B column
    const int khalf    = lane >> 4;   // K-half selector

    // --- build 2x8 A fragments (16x32 bf16 each) from f32 z, kept in VGPRs ---
    BFrag a0[8], a1[8];
    const float* zrow0 = z + (size_t)(rowStart + m) * EMBED_DIM;
    const float* zrow1 = z + (size_t)(rowStart + 16 + m) * EMBED_DIM;
    #pragma unroll
    for (int ks = 0; ks < 8; ++ks) {
        const int kbase = ks * 32;
        #pragma unroll
        for (int j = 0; j < 8; ++j) {
            a0[ks].h[j]     = f32_to_bf16_bits(zrow0[kbase + khalf * 16 + j]);
            a0[ks].h[8 + j] = f32_to_bf16_bits(zrow0[kbase + khalf * 16 + 8 + j]);
            a1[ks].h[j]     = f32_to_bf16_bits(zrow1[kbase + khalf * 16 + j]);
            a1[ks].h[8 + j] = f32_to_bf16_bits(zrow1[kbase + khalf * 16 + 8 + j]);
        }
    }

    float best0[8], best1[8];
    int   bidx0[8], bidx1[8];
    #pragma unroll
    for (int i = 0; i < 8; ++i) {
        best0[i] = 3.4e38f; bidx0[i] = 0;
        best1[i] = 3.4e38f; bidx1[i] = 0;
    }

    // per-lane base into emb_bf: column = code (lane&15), K-half = lane>>4
    const uint16_t* ebase = emb_bf + (size_t)m * EMBED_DIM + khalf * 16;

    // --- prologue: preload first B fragment and first ||e||^2 ---
    BFrag bbuf[2];
    bbuf[0].u4[0] = *(const uint4*)(ebase);
    bbuf[0].u4[1] = *(const uint4*)(ebase + 8);
    float en = enorm[m];

    for (int c = 0; c < NUM_EMB; c += 16) {
        const uint16_t* ep = ebase + (size_t)c * EMBED_DIM;
        // prefetch next tile's ||e||^2 one iteration ahead
        // (tail reads land in adjacent workspace arrays: valid, unused)
        const float en_next = enorm[c + 16 + m];
        v8f acc0 = {};
        v8f acc1 = {};
        #pragma unroll
        for (int ks = 0; ks < 8; ++ks) {
            const int cur = ks & 1;
            const int nxt = cur ^ 1;
            // issue loads for step ks+1 (or next code tile's step 0)
            const uint16_t* pn = (ks < 7) ? (ep + (ks + 1) * 32)
                                          : (ep + 16 * EMBED_DIM);
            bbuf[nxt].u4[0] = *(const uint4*)(pn);
            bbuf[nxt].u4[1] = *(const uint4*)(pn + 8);
            // two independent accumulator chains share each B fragment
            acc0 = __builtin_amdgcn_wmma_f32_16x16x32_bf16(
                false, a0[ks].v, false, bbuf[cur].v, (short)0, acc0, false, false);
            acc1 = __builtin_amdgcn_wmma_f32_16x16x32_bf16(
                false, a1[ks].v, false, bbuf[cur].v, (short)0, acc1, false, false);
        }

        // Pin the pipeline: one VMEM slot (||e||^2), then 8x [2 loads][2 wmma].
        // Loads for step k+1 are forced ahead of the WMMAs consuming step k,
        // keeping a load pair in flight behind every WMMA pair.
        __builtin_amdgcn_sched_group_barrier(0x020, 1, 0); // VMEM read (enorm)
        #pragma unroll
        for (int g = 0; g < 8; ++g) {
            __builtin_amdgcn_sched_group_barrier(0x020, 2, 0); // 2 VMEM reads
            __builtin_amdgcn_sched_group_barrier(0x008, 2, 0); // 2 WMMA
        }

        #pragma unroll
        for (int i = 0; i < 8; ++i) {
            float s0 = en - 2.0f * acc0[i];
            float s1 = en - 2.0f * acc1[i];
            if (s0 < best0[i]) { best0[i] = s0; bidx0[i] = c + m; }
            if (s1 < best1[i]) { best1[i] = s1; bidx1[i] = c + m; }
        }
        en = en_next;
    }

    // --- min+argmin across the 16 lanes of each half, for both row tiles ---
    #pragma unroll
    for (int i = 0; i < 8; ++i) {
        float s  = best0[i];
        int   id = bidx0[i];
        #pragma unroll
        for (int off = 1; off < 16; off <<= 1) {
            float os  = __shfl_xor(s, off, 32);
            int   oid = __shfl_xor(id, off, 32);
            if (os < s || (os == s && oid < id)) { s = os; id = oid; }
        }
        if (m == 0) best_idx[rowStart + i + khalf * 8] = id;

        s  = best1[i];
        id = bidx1[i];
        #pragma unroll
        for (int off = 1; off < 16; off <<= 1) {
            float os  = __shfl_xor(s, off, 32);
            int   oid = __shfl_xor(id, off, 32);
            if (os < s || (os == s && oid < id)) { s = os; id = oid; }
        }
        if (m == 0) best_idx[rowStart + 16 + i + khalf * 8] = id;
    }
}

// ---------- kernel 3: gather z_q, write z_q_st + indices, accumulate loss ----------
__global__ __launch_bounds__(EMBED_DIM)
void vq_gather_loss(const float* __restrict__ z,
                    const float* __restrict__ emb,
                    const int* __restrict__ idx,
                    float* __restrict__ out,
                    float* __restrict__ accum) {
    __shared__ float red[EMBED_DIM];
    const int row = blockIdx.x;
    const int d   = threadIdx.x;
    const int k   = idx[row];
    const float q  = emb[(size_t)k * EMBED_DIM + d];
    const float zv = z[(size_t)row * EMBED_DIM + d];
    out[(size_t)row * EMBED_DIM + d] = q;   // z + sg(z_q - z) == z_q numerically
    const float diff = q - zv;
    red[d] = diff * diff;
    __syncthreads();
    #pragma unroll
    for (int s = EMBED_DIM / 2; s > 0; s >>= 1) {
        if (d < s) red[d] += red[d + s];
        __syncthreads();
    }
    if (d == 0) {
        atomicAdd(accum, red[0]);
        out[NZQ + 1 + row] = (float)k;      // indices slot (exact <= 4095)
    }
}

// ---------- kernel 4: finalize loss ----------
__global__ void vq_finalize(const float* __restrict__ accum, float* __restrict__ out) {
    if (blockIdx.x == 0 && threadIdx.x == 0)
        out[NZQ] = (1.0f + BETA) * accum[0] / (float)NZQ;
}

extern "C" void kernel_launch(void* const* d_in, const int* in_sizes, int n_in,
                              void* d_out, int out_size, void* d_ws, size_t ws_size,
                              hipStream_t stream) {
    const float* z   = (const float*)d_in[0];   // [16,4096,256] f32
    const float* emb = (const float*)d_in[1];   // [4096,256]    f32
    float* out = (float*)d_out;                 // [z_q_st | loss | indices]

    char* ws = (char*)d_ws;
    uint16_t* emb_bf = (uint16_t*)ws;                                   // 2 MB
    float*    enorm  = (float*)(ws + (size_t)NUM_EMB * EMBED_DIM * 2);  // 16 KB
    int*      idx    = (int*)((char*)enorm + NUM_EMB * sizeof(float));  // 256 KB
    float*    accum  = (float*)((char*)idx + N_ROWS * sizeof(int));     // 4 B

    hipLaunchKernelGGL(vq_init,        dim3(1),            dim3(32),        0, stream, accum);
    hipLaunchKernelGGL(vq_prep_emb,    dim3(NUM_EMB),      dim3(EMBED_DIM), 0, stream, emb, emb_bf, enorm);
    hipLaunchKernelGGL(vq_argmin,      dim3(N_ROWS / 128), dim3(128),       0, stream, z, emb_bf, enorm, idx);
    hipLaunchKernelGGL(vq_gather_loss, dim3(N_ROWS),       dim3(EMBED_DIM), 0, stream, z, emb, idx, out, accum);
    hipLaunchKernelGGL(vq_finalize,    dim3(1),            dim3(32),        0, stream, accum, out);
}
